// MambaLayer_21466246545714
// MI455X (gfx1250) — matrix operand order repaired and checked by reading, not compiled, not run on hardware
//
#include <hip/hip_runtime.h>
#include <hip/hip_bf16.h>

typedef __attribute__((ext_vector_type(16))) _Float16 v16h;
typedef __attribute__((ext_vector_type(8)))  _Float16 v8h;
typedef __attribute__((ext_vector_type(8)))  float    v8f;

// ---------------------------------------------------------------------------
// LayerNorm over the feature dim with permuted gather from x(B,C,H,W).
// Writes fp16 rows (row-major (rows, dm)) for the WMMA in-proj GEMM.
// mode 0: l=h*W+w, feat=c   (dm=C)
// mode 1: l=c*W+w, feat=h   (dm=H)
// mode 2: l=c*H+h, feat=w   (dm=W)
// ---------------------------------------------------------------------------
__global__ void layernorm_gather(const float* __restrict__ x,
                                 const float* __restrict__ gw,
                                 const float* __restrict__ gb,
                                 _Float16* __restrict__ outh,
                                 int mode, int C, int H, int W,
                                 int L, int dm) {
  int row = blockIdx.x;               // b*L + l
  int bb  = row / L, l = row % L;
  int tid = threadIdx.x;              // 64 threads
  __shared__ float s1[64], s2[64];
  float v[4];
  int nv = 0;
  float sum = 0.f, sq = 0.f;
  for (int c = tid; c < dm; c += 64) {
    size_t src;
    if (mode == 0)      src = ((size_t)(bb * C + c)        * H + (l / W)) * W + (l % W);
    else if (mode == 1) src = ((size_t)(bb * C + (l / W))  * H + c)       * W + (l % W);
    else                src = ((size_t)(bb * C + (l / H))  * H + (l % H)) * W + c;
    float t = x[src];
    v[nv++] = t;
    sum += t; sq += t * t;
  }
  s1[tid] = sum; s2[tid] = sq;
  __syncthreads();
  for (int off = 32; off > 0; off >>= 1) {
    if (tid < off) { s1[tid] += s1[tid + off]; s2[tid] += s2[tid + off]; }
    __syncthreads();
  }
  float mu   = s1[0] / (float)dm;
  float var  = s2[0] / (float)dm - mu * mu;
  float rstd = rsqrtf(var + 1e-5f);
  nv = 0;
  for (int c = tid; c < dm; c += 64) {
    float t = (v[nv++] - mu) * rstd * gw[c] + gb[c];
    outh[(size_t)row * dm + c] = (_Float16)t;
  }
}

// fp32 -> fp16 with optional zero row-padding: dst is (rows_pad, cols)
__global__ void f2h_pad(const float* __restrict__ src, _Float16* __restrict__ dst,
                        int rows, int cols, int rows_pad) {
  size_t i   = (size_t)blockIdx.x * blockDim.x + threadIdx.x;
  size_t tot = (size_t)rows_pad * cols;
  if (i >= tot) return;
  int r = (int)(i / cols), c = (int)(i % cols);
  dst[i] = (r < rows) ? (_Float16)src[(size_t)r * cols + c] : (_Float16)0.f;
}

// ---------------------------------------------------------------------------
// WMMA fp16 GEMM:  C(M,N) = A(M,K) * W(N,K)^T, fp32 accumulate.
// One wave per 16x16 tile; K%32==0, M%16==0, N%16==0.
// Fragment layouts per CDNA5 ISA 7.12.2 (16-bit A 16x32, B 32x16).
// ---------------------------------------------------------------------------
__global__ void wmma_gemm_nt(const _Float16* __restrict__ A,
                             const _Float16* __restrict__ Wm,
                             float* __restrict__ C,
                             int M, int N, int K) {
  int wave   = blockIdx.x * (blockDim.x >> 5) + (threadIdx.x >> 5);
  int tilesN = N >> 4;
  int tm = (wave / tilesN) << 4;
  int tn = (wave % tilesN) << 4;
  if (tm >= M) return;                         // wave-uniform guard (EXEC stays all-1s)
  int lane = threadIdx.x & 31;
  int hlf  = lane >> 4;                        // 0 or 1
  int l16  = lane & 15;
  const _Float16* arow = A  + (size_t)(tm + l16) * K;
  const _Float16* wrow = Wm + (size_t)(tn + l16) * K;
  v8f acc = {};
  union AV { v16h v; v8h h[2]; };
  for (int kk = 0; kk < K; kk += 32) {
    AV a, b;
    // A: lanes<16 carry K={kk..kk+7, kk+16..kk+23}; lanes>=16 carry +8
    a.h[0] = *reinterpret_cast<const v8h*>(arow + kk + 8 * hlf);
    a.h[1] = *reinterpret_cast<const v8h*>(arow + kk + 16 + 8 * hlf);
    // B: lanes<16 carry K={kk..kk+7, kk+16..kk+23} of column n; lanes>=16 carry +8
    b.h[0] = *reinterpret_cast<const v8h*>(wrow + kk + 8 * hlf);
    b.h[1] = *reinterpret_cast<const v8h*>(wrow + kk + 16 + 8 * hlf);
    acc = __builtin_amdgcn_wmma_f32_16x16x32_f16(false, a.v, false, b.v,
                                                 (short)0, acc, false, false);
  }
  // D layout: VGPR i holds (M = tm + i + 8*hlf, N = tn + l16)
  float* crow = C + (size_t)(tm + 8 * hlf) * N + tn + l16;
#pragma unroll
  for (int i = 0; i < 8; ++i) crow[(size_t)i * N] = acc[i];
}

// Depthwise causal conv (d_conv=4) + bias + SiLU over xz[:, :din]; fp32+fp16 out
__global__ void conv_silu(const float* __restrict__ xz,   // (B*L, 2*din)
                          const float* __restrict__ cw,   // (din, 4)
                          const float* __restrict__ cb,
                          float* __restrict__ xc,         // (B*L, din)
                          _Float16* __restrict__ xch,
                          int L, int din, size_t tot) {
  size_t i = (size_t)blockIdx.x * blockDim.x + threadIdx.x;
  if (i >= tot) return;
  int d      = (int)(i % din);
  size_t rl  = i / din;
  int l      = (int)(rl % L);
  size_t b0  = (rl / L) * (size_t)L;
  int stride = 2 * din;
  float acc  = cb[d];
#pragma unroll
  for (int j = 0; j < 4; ++j) {
    int ls = l - 3 + j;
    if (ls >= 0) acc += cw[d * 4 + j] * xz[(b0 + ls) * stride + d];
  }
  float s = acc / (1.f + expf(-acc));
  xc[i]  = s;
  xch[i] = (_Float16)s;
}

// dt2 = softplus(x_dbl[:, :dtr] @ dt_w^T + dt_b)
__global__ void dtproj(const float* __restrict__ xdbl,  // (M, Npad)
                       const float* __restrict__ dtw,   // (din, dtr)
                       const float* __restrict__ dtb,
                       float* __restrict__ dt2,         // (M, din)
                       int din, int dtr, int Npad, size_t tot) {
  size_t i = (size_t)blockIdx.x * blockDim.x + threadIdx.x;
  if (i >= tot) return;
  int d      = (int)(i % din);
  size_t row = i / din;
  float acc  = dtb[d];
  for (int r = 0; r < dtr; ++r)
    acc += xdbl[row * Npad + r] * dtw[d * dtr + r];
  dt2[i] = (acc > 20.f) ? acc : log1pf(expf(acc));
}

// Selective scan: one 16-lane group per (b,d); lane = state index n.
// h = exp(dt*A)*h + dt*B*xc ; y = (sum_n h*C) + D*xc ; gated by silu(z).
__global__ void ssm_scan(const float* __restrict__ dt2,   // (B*L, din)
                         const float* __restrict__ xdbl,  // (B*L, Npad)
                         const float* __restrict__ xc,    // (B*L, din)
                         const float* __restrict__ xz,    // (B*L, 2*din)
                         const float* __restrict__ Alog,  // (din, 16)
                         const float* __restrict__ Dp,    // (din)
                         _Float16* __restrict__ yh,       // (B*L, din)
                         int L, int din, int dtr, int Npad, int ngrp) {
  int grp = (int)(((size_t)blockIdx.x * blockDim.x + threadIdx.x) >> 4);
  int n   = threadIdx.x & 15;
  if (grp >= ngrp) return;
  int bb = grp / din, d = grp % din;
  float A  = -expf(Alog[d * 16 + n]);
  float Dd = Dp[d];
  float h  = 0.f;
  size_t row0 = (size_t)bb * L;
  for (int l = 0; l < L; ++l) {
    size_t row = row0 + l;
    float dt  = dt2[row * din + d];
    float xcv = xc [row * din + d];
    float Bv  = xdbl[row * Npad + dtr + n];
    float Cv  = xdbl[row * Npad + dtr + 16 + n];
    h = expf(dt * A) * h + dt * Bv * xcv;
    float p = h * Cv;
    p += __shfl_xor(p, 8, 16);
    p += __shfl_xor(p, 4, 16);
    p += __shfl_xor(p, 2, 16);
    p += __shfl_xor(p, 1, 16);
    if (n == 0) {
      float zv = xz[row * (size_t)(2 * din) + din + d];
      float g  = zv / (1.f + expf(-zv));
      yh[row * din + d] = (_Float16)((p + Dd * xcv) * g);
    }
  }
}

// Scatter (b,l,m) -> (b,c,h,w) with weight 1/3; bijective per direction.
__global__ void accum_out(const float* __restrict__ src, float* __restrict__ out,
                          int mode, int first, int C, int H, int W,
                          int L, int dm, size_t tot) {
  size_t idx = (size_t)blockIdx.x * blockDim.x + threadIdx.x;
  if (idx >= tot) return;
  int m     = (int)(idx % dm);
  size_t rl = idx / dm;
  int l     = (int)(rl % L);
  int bb    = (int)(rl / L);
  size_t dst;
  if (mode == 0)      dst = ((size_t)(bb * C + m)       * H + (l / W)) * W + (l % W);
  else if (mode == 1) dst = ((size_t)(bb * C + (l / W)) * H + m)       * W + (l % W);
  else                dst = ((size_t)(bb * C + (l / H)) * H + (l % H)) * W + m;
  float v = src[idx] * (1.0f / 3.0f);
  if (first) out[dst] = v; else out[dst] += v;
}

// ---------------------------------------------------------------------------
extern "C" void kernel_launch(void* const* d_in, const int* in_sizes, int n_in,
                              void* d_out, int out_size, void* d_ws, size_t ws_size,
                              hipStream_t stream) {
  (void)in_sizes; (void)n_in; (void)out_size; (void)ws_size;
  const float* x       = (const float*)d_in[0];
  const float* norm_w  = (const float*)d_in[1];
  const float* norm_b  = (const float*)d_in[2];
  const float* norm2_w = (const float*)d_in[3];
  const float* norm2_b = (const float*)d_in[4];
  struct MP { const float *in_w, *conv_w, *conv_b, *xproj_w, *dt_w, *dt_b, *A_log, *D, *out_w; };
  MP m1 = { (const float*)d_in[5],  (const float*)d_in[6],  (const float*)d_in[7],
            (const float*)d_in[8],  (const float*)d_in[9],  (const float*)d_in[10],
            (const float*)d_in[11], (const float*)d_in[12], (const float*)d_in[13] };
  MP m2 = { (const float*)d_in[14], (const float*)d_in[15], (const float*)d_in[16],
            (const float*)d_in[17], (const float*)d_in[18], (const float*)d_in[19],
            (const float*)d_in[20], (const float*)d_in[21], (const float*)d_in[22] };

  const int Bsz = 2, C = 192, H = 64, W = 64;
  const int Npad = 48;

  // --- carve workspace (max sizes across directions; reused sequentially) ---
  char* wp = (char*)d_ws;
  auto carve = [&](size_t bytes) { char* p = wp; wp += (bytes + 255) & ~(size_t)255; return p; };
  const size_t MDM  = 1572864;   // max M*dm   (8192*192 == 24576*64)
  const size_t MDI  = 3145728;   // max M*din  (8192*384 == 24576*128)
  const size_t MXZ  = 6291456;   // max M*2*din
  const size_t MXD  = 24576ull * Npad;
  _Float16* xnh     = (_Float16*)carve(MDM * 2);
  float*    xz      = (float*)   carve(MXZ * 4);
  float*    xc      = (float*)   carve(MDI * 4);
  _Float16* xch     = (_Float16*)carve(MDI * 2);
  float*    xdbl    = (float*)   carve(MXD * 4);
  float*    dt2     = (float*)   carve(MDI * 4);
  _Float16* yh      = (_Float16*)carve(MDI * 2);
  float*    outbuf  = (float*)   carve(MDM * 4);
  _Float16* inwh    = (_Float16*)carve(147456 * 2);
  _Float16* xprojwh = (_Float16*)carve((size_t)Npad * 384 * 2);
  _Float16* outwh   = (_Float16*)carve(73728 * 2);

  for (int dir = 0; dir < 3; ++dir) {
    const MP& P = (dir == 0) ? m1 : m2;
    int dm  = (dir == 0) ? 192 : 64;
    int L   = (dir == 0) ? H * W : (dir == 1 ? C * W : C * H);
    int din = 2 * dm;                  // expand = 2
    int dtr = (dm + 15) / 16;          // ceil(dm/16): 12 or 4
    int M   = Bsz * L;
    const float* gw = (dir == 0) ? norm_w : norm2_w;
    const float* gb = (dir == 0) ? norm_b : norm2_b;

    // 1) layernorm + permuted gather -> fp16 rows
    layernorm_gather<<<M, 64, 0, stream>>>(x, gw, gb, xnh, dir, C, H, W, L, dm);

    // 2) weight conversions (fp32 -> fp16, x-proj zero-padded to 48 rows)
    {
      size_t t1 = (size_t)(2 * din) * dm;
      f2h_pad<<<(int)((t1 + 255) / 256), 256, 0, stream>>>(P.in_w, inwh, 2 * din, dm, 2 * din);
      size_t t2 = (size_t)Npad * din;
      f2h_pad<<<(int)((t2 + 255) / 256), 256, 0, stream>>>(P.xproj_w, xprojwh, dtr + 32, din, Npad);
      size_t t3 = (size_t)dm * din;
      f2h_pad<<<(int)((t3 + 255) / 256), 256, 0, stream>>>(P.out_w, outwh, dm, din, dm);
    }

    // 3) in-proj: (M,dm) x (2*din,dm)^T -> xz (M, 2*din)     [WMMA]
    {
      int tiles = (M / 16) * ((2 * din) / 16);
      wmma_gemm_nt<<<(tiles + 7) / 8, 256, 0, stream>>>(xnh, inwh, xz, M, 2 * din, dm);
    }

    // 4) depthwise causal conv + SiLU
    {
      size_t tot = (size_t)M * din;
      conv_silu<<<(int)((tot + 255) / 256), 256, 0, stream>>>(xz, P.conv_w, P.conv_b,
                                                              xc, xch, L, din, tot);
    }

    // 5) x-proj: (M,din) x (48,din)^T -> xdbl (M,48)          [WMMA]
    {
      int tiles = (M / 16) * (Npad / 16);
      wmma_gemm_nt<<<(tiles + 7) / 8, 256, 0, stream>>>(xch, xprojwh, xdbl, M, Npad, din);
    }

    // 6) dt projection + softplus (K = 12 or 4: scalar)
    {
      size_t tot = (size_t)M * din;
      dtproj<<<(int)((tot + 255) / 256), 256, 0, stream>>>(xdbl, P.dt_w, P.dt_b,
                                                           dt2, din, dtr, Npad, tot);
    }

    // 7) selective scan (sequential in L; parallel over B*din groups of 16 lanes)
    {
      int ngrp = Bsz * din;                         // 768 or 256 (multiple of 16)
      ssm_scan<<<ngrp / 16, 256, 0, stream>>>(dt2, xdbl, xc, xz, P.A_log, P.D,
                                              yh, L, din, dtr, Npad, ngrp);
    }

    // 8) out-proj: (M,din) x (dm,din)^T -> outbuf (M,dm)      [WMMA]
    {
      int tiles = (M / 16) * (dm / 16);
      wmma_gemm_nt<<<(tiles + 7) / 8, 256, 0, stream>>>(yh, outwh, outbuf, M, dm, din);
    }

    // 9) permuted accumulate into d_out with weight 1/3
    {
      size_t tot = (size_t)M * dm;
      accum_out<<<(int)((tot + 255) / 256), 256, 0, stream>>>(outbuf, (float*)d_out,
                                                              dir, dir == 0 ? 1 : 0,
                                                              C, H, W, L, dm, tot);
    }
  }
}
